// CustomMultiheadAttention_2439541424616
// MI455X (gfx1250) — matrix operand order repaired and checked by reading, not compiled, not run on hardware
//
#include <hip/hip_runtime.h>
#include <hip/hip_bf16.h>
#include <stdint.h>

// ---------------------------------------------------------------------------
// Multi-head attention (B=4, H=16, S=2048, E=1024, D=64) for gfx1250.
// All GEMMs via v_wmma_f32_16x16x32_bf16 (wave32); operand staging through
// LDS with GLOBAL_LOAD_ASYNC_TO_LDS_B128 double buffering (ASYNCcnt).
// ---------------------------------------------------------------------------

#define EMB   1024
#define NH    16
#define HD    64
#define BATCH 4
#define SEQ   2048
#define MROWS (BATCH * SEQ)      // 8192

typedef __attribute__((ext_vector_type(16))) __bf16 v16bf;
typedef __attribute__((ext_vector_type(8)))  __bf16 v8bf;
typedef __attribute__((ext_vector_type(8)))  float  v8f;

// float -> bf16 (round to nearest even), stored as ushort
static __device__ __forceinline__ unsigned short f2bf(float f) {
    uint32_t u = __float_as_uint(f);
    u += 0x7FFFu + ((u >> 16) & 1u);
    return (unsigned short)(u >> 16);
}

static __device__ __forceinline__ v8f wmma_bf16(v16bf a, v16bf b, v8f c) {
    // (neg_a, A, neg_b, B, c_mod, C, reuse_a, reuse_b)
    return __builtin_amdgcn_wmma_f32_16x16x32_bf16(false, a, false, b,
                                                   (short)0, c, false, false);
}

// A-fragment (16x32 bf16, M = lane%16).  Per ISA 7.12.2: lane L holds
// K = 8*(L>>4)+[0..7] in VGPR0-3 and K = 16+8*(L>>4)+[0..7] in VGPR4-7.
static __device__ __forceinline__ v16bf load_afrag(const unsigned short* base, int hi) {
    v8bf a0 = *(const v8bf*)(base + 8 * hi);
    v8bf a1 = *(const v8bf*)(base + 16 + 8 * hi);
    return __builtin_shufflevector(a0, a1, 0, 1, 2, 3, 4, 5, 6, 7,
                                           8, 9, 10, 11, 12, 13, 14, 15);
}

// B-fragment (32x16 bf16, N = lane%16): lane L holds K = 8*(L>>4)+[0..7]
// and K = 16+8*(L>>4)+[0..7] (same order as A).  `colbase` = column-major row start + k0.
static __device__ __forceinline__ v16bf load_bfrag(const unsigned short* colbase, int hi) {
    v8bf b0 = *(const v8bf*)(colbase + 8 * hi);
    v8bf b1 = *(const v8bf*)(colbase + 16 + 8 * hi);
    return __builtin_shufflevector(b0, b1, 0, 1, 2, 3, 4, 5, 6, 7,
                                           8, 9, 10, 11, 12, 13, 14, 15);
}

// Async copy of 16B (per lane) from global to LDS; tracked by ASYNCcnt.
static __device__ __forceinline__ void async_b128(const unsigned short* gp, void* lp) {
    unsigned lds  = (unsigned)(uintptr_t)lp;          // addr[31:0] == LDS offset
    uint64_t gadr = (uint64_t)(uintptr_t)gp;
    asm volatile("global_load_async_to_lds_b128 %0, %1, off"
                 :: "v"(lds), "v"(gadr) : "memory");
}
#define WAIT_ASYNC(imm) asm volatile("s_wait_asynccnt " #imm ::: "memory")

// ---------------------------------------------------------------------------
// Stage 1a: fp32 -> bf16 elementwise
// ---------------------------------------------------------------------------
__global__ void k_cvt_bf16(const float* __restrict__ src,
                           unsigned short* __restrict__ dst, int n) {
    int i = blockIdx.x * blockDim.x + threadIdx.x;
    if (i < n) dst[i] = f2bf(src[i]);
}

// Stage 1b: fp32 W[k][n] -> bf16 Wt[n][k]  (B-fragment friendly)
__global__ void k_wtrans(const float* __restrict__ W,
                         unsigned short* __restrict__ Wt) {
    int i = blockIdx.x * blockDim.x + threadIdx.x;   // i over E*E
    int n = i & (EMB - 1);
    int k = i >> 10;
    Wt[(size_t)n * EMB + k] = f2bf(W[(size_t)k * EMB + n]);
}

// ---------------------------------------------------------------------------
// Stage 2/4: GEMM  C[M,N] = A[M,K]bf16 * Wt[N,K]bf16 + bias
//   mode 0: store bf16 head-split      [B,H,S,D]
//   mode 1: store bf16 head-split + T  [B,H,D,S]   (for V)
//   mode 2: store fp32 plain           [M,N]       (final output)
// One wave -> 16(M) x 64(N); block = 8 waves stacked in M (128x64 tile).
// B tile (64 cols x 32 K = 4KB) staged in LDS via async copy, double-buffered
// and shared by all 8 waves; A fragments software-pipelined in registers.
// ---------------------------------------------------------------------------
__global__ void __launch_bounds__(256)
k_gemm_bias(const unsigned short* __restrict__ A,
            const unsigned short* __restrict__ Wt,
            const float* __restrict__ bias,
            unsigned short* __restrict__ out_bf,
            float* __restrict__ out_f,
            int mode) {
    __shared__ unsigned short Bl[2][64][32];   // [buf][n][k]  4KB each

    const int tid    = threadIdx.x;
    const int w      = tid >> 5;
    const int lane   = tid & 31;
    const int lane_m = lane & 15;
    const int hi     = lane >> 4;
    const int m0     = blockIdx.x * 128 + w * 16;
    const int n0     = blockIdx.y * 64;

    // cooperative fill: thread t copies 16B; col = t>>2, chunk = (t&3)*8
    const int f_col = tid >> 2;
    const int f_ch  = (tid & 3) * 8;
    const unsigned short* WtRow = Wt + (size_t)(n0 + f_col) * EMB + f_ch;

    const unsigned short* Arow = A + (size_t)(m0 + lane_m) * EMB;

    v8f acc0 = {}, acc1 = {}, acc2 = {}, acc3 = {};

    async_b128(WtRow, &Bl[0][f_col][f_ch]);            // prefill buffer 0
    v16bf a_cur = load_afrag(Arow, hi);

    for (int k0 = 0; k0 < EMB; k0 += 32) {
        const int  cur  = (k0 >> 5) & 1;
        const bool more = (k0 + 32) < EMB;
        v16bf a_nxt = a_cur;
        if (more) {
            async_b128(WtRow + k0 + 32, &Bl[cur ^ 1][f_col][f_ch]);
            a_nxt = load_afrag(Arow + k0 + 32, hi);
            WAIT_ASYNC(0x1);                           // older fill done
        } else {
            WAIT_ASYNC(0x0);
        }
        __syncthreads();                               // tile visible block-wide

        v16bf b0 = load_bfrag(&Bl[cur][ 0 + lane_m][0], hi);
        v16bf b1 = load_bfrag(&Bl[cur][16 + lane_m][0], hi);
        v16bf b2 = load_bfrag(&Bl[cur][32 + lane_m][0], hi);
        v16bf b3 = load_bfrag(&Bl[cur][48 + lane_m][0], hi);
        acc0 = wmma_bf16(a_cur, b0, acc0);
        acc1 = wmma_bf16(a_cur, b1, acc1);
        acc2 = wmma_bf16(a_cur, b2, acc2);
        acc3 = wmma_bf16(a_cur, b3, acc3);

        __syncthreads();                               // reads done before refill
        a_cur = a_nxt;
    }

    v8f accs[4] = {acc0, acc1, acc2, acc3};
    #pragma unroll
    for (int t = 0; t < 4; ++t) {
        #pragma unroll
        for (int i = 0; i < 8; ++i) {
            int m = m0 + i + 8 * hi;
            int n = n0 + t * 16 + lane_m;
            float v = accs[t][i] + bias[n];
            if (mode == 2) {
                out_f[(size_t)m * EMB + n] = v;
            } else {
                int b = m >> 11, s = m & (SEQ - 1);
                int h = n >> 6,  d = n & (HD - 1);
                size_t idx = (mode == 0)
                    ? ((size_t)((b * NH + h) * SEQ + s) * HD + d)     // [B,H,S,D]
                    : ((size_t)((b * NH + h) * HD  + d) * SEQ + s);   // [B,H,D,S]
                out_bf[idx] = f2bf(v);
            }
        }
    }
}

// ---------------------------------------------------------------------------
// Stage 3: flash attention.  One wave per (b, h, 16-query tile); the 8 waves
// of a block share one (b,h), so K/V 32-key tiles are staged in LDS with
// async double buffering and consumed by all waves.
// ---------------------------------------------------------------------------
__global__ void __launch_bounds__(256)
k_attention(const unsigned short* __restrict__ Qh,   // [B,H,S,D] bf16
            const unsigned short* __restrict__ Kh,   // [B,H,S,D] bf16
            const unsigned short* __restrict__ Vt,   // [B,H,D,S] bf16
            unsigned short* __restrict__ attn) {     // [B,S,E]   bf16
    __shared__ unsigned short Kl[2][32][64];         // [buf][key][d]   4KB each
    __shared__ unsigned short Vl[2][64][32];         // [buf][d][key]   4KB each
    __shared__ float          sc_lds[8][16][32];
    __shared__ unsigned short p_lds [8][16][32];
    __shared__ float          m_lds [8][16];
    __shared__ float          l_lds [8][16];
    __shared__ float          a_lds [8][16];

    const int tid    = threadIdx.x;
    const int w      = tid >> 5;
    const int lane   = tid & 31;
    const int lane_m = lane & 15;
    const int hi     = lane >> 4;

    const int g  = blockIdx.x * 8 + w;      // global wave id
    const int qt = g & 127;                 // query tile (S/16)
    const int bh = blockIdx.x >> 4;         // uniform: 8 waves share (b,h)
    const int b  = bh >> 4, h = bh & 15;

    const unsigned short* Qp = Qh + ((size_t)bh * SEQ + qt * 16) * HD;
    const unsigned short* Kb = Kh + (size_t)bh * SEQ * HD;
    const unsigned short* Vp = Vt + (size_t)bh * HD * SEQ;

    // K fill: thread t -> row t>>3 (32 keys), chunk (t&7)*8 of 64 d's
    const int kf_r = tid >> 3, kf_c = (tid & 7) * 8;
    // V fill: thread t -> row t>>2 (64 d's),  chunk (t&3)*8 of 32 keys
    const int vf_r = tid >> 2, vf_c = (tid & 3) * 8;

    // Q fragments, reused across all key blocks (D = 64 -> two K=32 chunks)
    v16bf aq0 = load_afrag(Qp + lane_m * HD +  0, hi);
    v16bf aq1 = load_afrag(Qp + lane_m * HD + 32, hi);

    if (lane < 16) {
        m_lds[w][lane] = -INFINITY;
        l_lds[w][lane] = 0.0f;
    }
    v8f o0 = {}, o1 = {}, o2 = {}, o3 = {};
    const float scale = 0.125f;             // 1/sqrt(64)

    // prefill buffer 0 with key block 0
    async_b128(Kb + (size_t)kf_r * HD + kf_c,  &Kl[0][kf_r][kf_c]);
    async_b128(Vp + (size_t)vf_r * SEQ + vf_c, &Vl[0][vf_r][vf_c]);

    for (int kb = 0; kb < SEQ / 32; ++kb) {
        const int  cur  = kb & 1;
        const bool more = (kb + 1) < SEQ / 32;
        if (more) {
            async_b128(Kb + (size_t)((kb + 1) * 32 + kf_r) * HD + kf_c,
                       &Kl[cur ^ 1][kf_r][kf_c]);
            async_b128(Vp + (size_t)vf_r * SEQ + (kb + 1) * 32 + vf_c,
                       &Vl[cur ^ 1][vf_r][vf_c]);
            WAIT_ASYNC(0x2);                // current tiles complete
        } else {
            WAIT_ASYNC(0x0);
        }
        __syncthreads();

        // ---- scores 16x32 = Q(16x64) . K^T (B-frag column n = key) ----
        v8f s0 = {}, s1 = {};
        {
            v16bf bk;
            bk = load_bfrag(&Kl[cur][ 0 + lane_m][ 0], hi); s0 = wmma_bf16(aq0, bk, s0);
            bk = load_bfrag(&Kl[cur][ 0 + lane_m][32], hi); s0 = wmma_bf16(aq1, bk, s0);
            bk = load_bfrag(&Kl[cur][16 + lane_m][ 0], hi); s1 = wmma_bf16(aq0, bk, s1);
            bk = load_bfrag(&Kl[cur][16 + lane_m][32], hi); s1 = wmma_bf16(aq1, bk, s1);
        }
        // C layout -> LDS rows (row = i + 8*hi, col = lane_m)
        #pragma unroll
        for (int i = 0; i < 8; ++i) {
            sc_lds[w][i + 8 * hi][ 0 + lane_m] = s0[i] * scale;
            sc_lds[w][i + 8 * hi][16 + lane_m] = s1[i] * scale;
        }
        __syncthreads();

        // ---- online softmax: lane r < 16 owns query row r ----
        if (lane < 16) {
            float mold = m_lds[w][lane];
            float mx = mold;
            #pragma unroll
            for (int j = 0; j < 32; ++j) mx = fmaxf(mx, sc_lds[w][lane][j]);
            float al  = __expf(mold - mx);
            float sum = 0.0f;
            #pragma unroll
            for (int j = 0; j < 32; ++j) {
                float p = __expf(sc_lds[w][lane][j] - mx);
                p_lds[w][lane][j] = f2bf(p);
                sum += p;
            }
            m_lds[w][lane] = mx;
            l_lds[w][lane] = l_lds[w][lane] * al + sum;
            a_lds[w][lane] = al;
        }
        __syncthreads();

        // ---- rescale accumulators by alpha(row) ----
        #pragma unroll
        for (int i = 0; i < 8; ++i) {
            float al = a_lds[w][i + 8 * hi];
            o0[i] *= al; o1[i] *= al; o2[i] *= al; o3[i] *= al;
        }

        // ---- O += P(16x32) . V(32x64) ----
        v16bf pa = load_afrag(&p_lds[w][lane_m][0], hi);
        v16bf bv;
        bv = load_bfrag(&Vl[cur][ 0 + lane_m][0], hi); o0 = wmma_bf16(pa, bv, o0);
        bv = load_bfrag(&Vl[cur][16 + lane_m][0], hi); o1 = wmma_bf16(pa, bv, o1);
        bv = load_bfrag(&Vl[cur][32 + lane_m][0], hi); o2 = wmma_bf16(pa, bv, o2);
        bv = load_bfrag(&Vl[cur][48 + lane_m][0], hi); o3 = wmma_bf16(pa, bv, o3);

        __syncthreads();                    // all reads of cur done before refill
    }

    // ---- normalize by l and store merged-head bf16 [B,S,E] ----
    #pragma unroll
    for (int i = 0; i < 8; ++i) {
        int   row = i + 8 * hi;
        float inv = 1.0f / l_lds[w][row];
        int   q   = qt * 16 + row;
        size_t base = ((size_t)(b * SEQ + q)) * EMB + h * HD;
        attn[base +  0 + lane_m] = f2bf(o0[i] * inv);
        attn[base + 16 + lane_m] = f2bf(o1[i] * inv);
        attn[base + 32 + lane_m] = f2bf(o2[i] * inv);
        attn[base + 48 + lane_m] = f2bf(o3[i] * inv);
    }
}

// ---------------------------------------------------------------------------
// Host-side launch
// ---------------------------------------------------------------------------
extern "C" void kernel_launch(void* const* d_in, const int* in_sizes, int n_in,
                              void* d_out, int out_size, void* d_ws, size_t ws_size,
                              hipStream_t stream) {
    const float* query = (const float*)d_in[0];
    const float* key   = (const float*)d_in[1];
    const float* value = (const float*)d_in[2];
    const float* Wq = (const float*)d_in[3];
    const float* bq = (const float*)d_in[4];
    const float* Wk = (const float*)d_in[5];
    const float* bk = (const float*)d_in[6];
    const float* Wv = (const float*)d_in[7];
    const float* bv = (const float*)d_in[8];
    const float* Wo = (const float*)d_in[9];
    const float* bo = (const float*)d_in[10];
    float* out = (float*)d_out;

    const size_t ACT = (size_t)MROWS * EMB;
    const size_t WSZ = (size_t)EMB * EMB;

    char* ws = (char*)d_ws;
    size_t off = 0;
    auto take = [&](size_t elems) { void* p = ws + off; off += elems * 2; return p; };

    unsigned short* qx  = (unsigned short*)take(ACT);
    unsigned short* kx  = (unsigned short*)take(ACT);
    unsigned short* vx  = (unsigned short*)take(ACT);
    unsigned short* wqt = (unsigned short*)take(WSZ);
    unsigned short* wkt = (unsigned short*)take(WSZ);
    unsigned short* wvt = (unsigned short*)take(WSZ);
    unsigned short* wot = (unsigned short*)take(WSZ);
    unsigned short* Qh  = (unsigned short*)take(ACT);  // [B,H,S,D]
    unsigned short* Kh  = (unsigned short*)take(ACT);  // [B,H,S,D]
    unsigned short* Vt  = (unsigned short*)take(ACT);  // [B,H,D,S]
    unsigned short* att = (unsigned short*)take(ACT);  // [B,S,E]

    {
        int n = (int)ACT;
        dim3 grid((n + 255) / 256);
        k_cvt_bf16<<<grid, 256, 0, stream>>>(query, qx, n);
        k_cvt_bf16<<<grid, 256, 0, stream>>>(key,   kx, n);
        k_cvt_bf16<<<grid, 256, 0, stream>>>(value, vx, n);
        dim3 wgrid((unsigned)(WSZ / 256));
        k_wtrans<<<wgrid, 256, 0, stream>>>(Wq, wqt);
        k_wtrans<<<wgrid, 256, 0, stream>>>(Wk, wkt);
        k_wtrans<<<wgrid, 256, 0, stream>>>(Wv, wvt);
        k_wtrans<<<wgrid, 256, 0, stream>>>(Wo, wot);
    }
    {
        dim3 grid(MROWS / 128, EMB / 64);
        k_gemm_bias<<<grid, 256, 0, stream>>>(qx, wqt, bq, Qh, nullptr, 0);
        k_gemm_bias<<<grid, 256, 0, stream>>>(kx, wkt, bk, Kh, nullptr, 0);
        k_gemm_bias<<<grid, 256, 0, stream>>>(vx, wvt, bv, Vt, nullptr, 1);
    }
    {
        dim3 grid((BATCH * NH * (SEQ / 16)) / 8);
        k_attention<<<grid, 256, 0, stream>>>(Qh, Kh, Vt, att);
    }
    {
        dim3 grid(MROWS / 128, EMB / 64);
        k_gemm_bias<<<grid, 256, 0, stream>>>(att, wot, bo, nullptr, out, 2);
    }
}
